// LSTMDecoder_90735479095499
// MI455X (gfx1250) — matrix operand order repaired and checked by reading, not compiled, not run on hardware
//
#include <hip/hip_runtime.h>

#define BS_   128
#define LAT_  512
#define HID_  1024

typedef __attribute__((ext_vector_type(16))) __bf16 v16bf;
typedef __attribute__((ext_vector_type(8)))  __bf16 v8bf;
typedef __attribute__((ext_vector_type(8)))  float  v8f;

__device__ __forceinline__ unsigned short f2bf(float f) {
  union { float f; unsigned u; } c; c.f = f;
  unsigned r = c.u + 0x7FFFu + ((c.u >> 16) & 1u);
  return (unsigned short)(r >> 16);
}

__device__ __forceinline__ float sigm(float x) {
  return __builtin_amdgcn_rcpf(1.0f + __expf(-x));
}
__device__ __forceinline__ float tanh_f(float x) {
  return 1.0f - 2.0f * __builtin_amdgcn_rcpf(__expf(2.0f * x) + 1.0f);
}
__device__ __forceinline__ v8f splat8(float v) {
  v8f r;
#pragma unroll
  for (int i = 0; i < 8; ++i) r[i] = v;
  return r;
}

// A operand (16xK row-major bf16 in LDS) -> WMMA 16-bit A fragment:
// lane L<16 holds row L, K {k0..k0+7, k0+16..k0+23}; lane L>=16 the +8 shifted set.
__device__ __forceinline__ v16bf load_a_lds(const unsigned short* base, int stride,
                                            int k0, int lane) {
  int m = lane & 15;
  int koff = k0 + ((lane >> 4) << 3);
  v8bf lo = *(const v8bf*)(base + m * stride + koff);
  v8bf hi = *(const v8bf*)(base + m * stride + koff + 16);
  return __builtin_shufflevector(lo, hi, 0,1,2,3,4,5,6,7,8,9,10,11,12,13,14,15);
}

// B operand: pre-packed fragments, 1024 bf16 per (ntile,ktile) fragment, lane-major.
__device__ __forceinline__ v16bf load_b_pk(const unsigned short* __restrict__ pk,
                                           int frag, int lane) {
  return *(const v16bf*)(pk + ((size_t)frag << 10) + (lane << 4));
}

#define WMMA_BF16(a, b, c) \
  __builtin_amdgcn_wmma_f32_16x16x32_bf16(false, (a), false, (b), (short)0, (c), false, false)

// ---- one-time fp32 -> bf16 repack into WMMA B-fragment layout ----
// W is (N x K) row-major. Fragment (nt,kt): lane L holds n = nt*16+(L&15),
// k = kt*32 + (L>=16)*8 + e (e<8) or kt*32 + 16 + (L>=16)*8 + (e-8) (e>=8), stored contiguously.
__global__ void pack_w(const float* __restrict__ W, unsigned short* __restrict__ P,
                       int N, int K) {
  int tid = blockIdx.x * blockDim.x + threadIdx.x;
  int total = (N * K) >> 4;
  if (tid >= total) return;
  int frag = tid >> 5;
  int lane = tid & 31;
  int ktiles = K >> 5;
  int nt = frag / ktiles;
  int kt = frag - nt * ktiles;
  int n = nt * 16 + (lane & 15);
  int k = kt * 32 + ((lane >> 4) << 3);
  const float* src = W + (size_t)n * K + k;
  unsigned short* dst = P + ((size_t)frag << 10) + (lane << 4);
#pragma unroll
  for (int e = 0; e < 16; ++e) dst[e] = f2bf(src[e + ((e >> 3) << 3)]);
}

// ---- main kernel: 8 workgroups (16 batch rows each), 16 waves, full T loop ----
__global__ __launch_bounds__(512, 1) void lstm_dec(
    const float* __restrict__ x0,
    const float* __restrict__ b_f, const float* __restrict__ b_b,
    const float* __restrict__ b_out, const float* __restrict__ W_fc1,
    const float* __restrict__ b_fc1,
    const unsigned short* __restrict__ pk_ih_f,
    const unsigned short* __restrict__ pk_hh_f,
    const unsigned short* __restrict__ pk_ih_b,
    const unsigned short* __restrict__ pk_hh_b,
    const unsigned short* __restrict__ pk_out,
    float* __restrict__ out, int T) {
  __shared__ unsigned short x_lds[16 * LAT_];          // 16 KB  bf16 x_t
  __shared__ unsigned short h_lds[2][2][16 * HID_];    // 128 KB [dir][buf] bf16 h
  __shared__ float          c_lds[2][16 * HID_];       // 128 KB fp32 cell state
  __shared__ float          out_acc[16];

  const int tid  = threadIdx.x;
  const int lane = tid & 31;
  const int w    = tid >> 5;      // wave 0..15
  const int dir  = w >> 3;        // 0 = fwd, 1 = bwd
  const int jw   = w & 7;         // 8 waves per direction
  const int nloc = lane & 15;
  const int hi16 = lane >> 4;
  const int b0   = blockIdx.x * 16;

  // init LDS state
  for (int i = tid; i < 16 * HID_; i += 512) {
    h_lds[0][0][i] = 0; h_lds[0][1][i] = 0;
    h_lds[1][0][i] = 0; h_lds[1][1][i] = 0;
    c_lds[0][i] = 0.0f; c_lds[1][i] = 0.0f;
  }
  for (int i = tid; i < 16 * LAT_; i += 512)
    x_lds[i] = f2bf(x0[(size_t)(b0 + (i >> 9)) * LAT_ + (i & 511)]);
  if (tid < 16) out_acc[tid] = 0.0f;

  const float* bias = dir ? b_b : b_f;
  const unsigned short* pk_ih = dir ? pk_ih_b : pk_ih_f;
  const unsigned short* pk_hh = dir ? pk_hh_b : pk_hh_f;
  const float bo0  = b_out[w * 32 + nloc];
  const float bo1  = b_out[w * 32 + 16 + nloc];
  const float bfc1 = b_fc1[0];

  __syncthreads();

  int buf = 0;
  for (int t = 0; t < T; ++t) {
    const unsigned short* hsrc = h_lds[dir][buf];
    unsigned short*       hdst = h_lds[dir][buf ^ 1];

    float pacc[8];
#pragma unroll
    for (int r = 0; r < 8; ++r) pacc[r] = 0.0f;

#pragma unroll 1
    for (int jt = 0; jt < 8; ++jt) {
      const int jtg  = jw * 8 + jt;         // hid column tile 0..63
      const int ncol = jtg * 16 + nloc;     // this lane's hid column
      v8f zi = splat8(bias[ncol]);
      v8f zf = splat8(bias[HID_ + ncol]);
      v8f zg = splat8(bias[2 * HID_ + ncol]);
      v8f zo = splat8(bias[3 * HID_ + ncol]);

      // x @ Wih^T contribution, K = 512
#pragma unroll 4
      for (int kt = 0; kt < 16; ++kt) {
        v16bf a = load_a_lds(x_lds, LAT_, kt * 32, lane);
        zi = WMMA_BF16(a, load_b_pk(pk_ih, (jtg)       * 16 + kt, lane), zi);
        zf = WMMA_BF16(a, load_b_pk(pk_ih, (64  + jtg) * 16 + kt, lane), zf);
        zg = WMMA_BF16(a, load_b_pk(pk_ih, (128 + jtg) * 16 + kt, lane), zg);
        zo = WMMA_BF16(a, load_b_pk(pk_ih, (192 + jtg) * 16 + kt, lane), zo);
      }
      // h @ Whh^T contribution, K = 1024
#pragma unroll 4
      for (int kt = 0; kt < 32; ++kt) {
        v16bf a = load_a_lds(hsrc, HID_, kt * 32, lane);
        zi = WMMA_BF16(a, load_b_pk(pk_hh, (jtg)       * 32 + kt, lane), zi);
        zf = WMMA_BF16(a, load_b_pk(pk_hh, (64  + jtg) * 32 + kt, lane), zf);
        zg = WMMA_BF16(a, load_b_pk(pk_hh, (128 + jtg) * 32 + kt, lane), zg);
        zo = WMMA_BF16(a, load_b_pk(pk_hh, (192 + jtg) * 32 + kt, lane), zo);
      }

      const float wf = W_fc1[dir * HID_ + ncol];
#pragma unroll
      for (int r = 0; r < 8; ++r) {
        const int m = r + (hi16 << 3);
        float* cp = &c_lds[dir][m * HID_ + ncol];
        float c2 = sigm(zf[r]) * (*cp) + sigm(zi[r]) * tanh_f(zg[r]);
        *cp = c2;
        float h2 = sigm(zo[r]) * tanh_f(c2);
        hdst[m * HID_ + ncol] = f2bf(h2);
        pacc[r] += fmaxf(h2, 0.0f) * wf;       // relu(out_t) . W_fc1 partial
      }
    }

    // fc1 partial reduction across the 16 lanes of each half-wave
#pragma unroll
    for (int r = 0; r < 8; ++r) {
      float v = pacc[r];
      v += __shfl_xor(v, 1, 32);
      v += __shfl_xor(v, 2, 32);
      v += __shfl_xor(v, 4, 32);
      v += __shfl_xor(v, 8, 32);
      if (nloc == 0) atomicAdd(&out_acc[r + (hi16 << 3)], v);
    }

    __syncthreads();  // h[buf^1]/out_acc complete; all reads of x/h[buf] done

    // x_next = hb_new @ W_out^T + b_out  (all 16 waves, 2 n-tiles each)
    {
      const unsigned short* hb = h_lds[1][buf ^ 1];
      v8f a0 = splat8(bo0), a1 = splat8(bo1);
#pragma unroll 4
      for (int kt = 0; kt < 32; ++kt) {
        v16bf a = load_a_lds(hb, HID_, kt * 32, lane);
        a0 = WMMA_BF16(a, load_b_pk(pk_out, (2 * w)     * 32 + kt, lane), a0);
        a1 = WMMA_BF16(a, load_b_pk(pk_out, (2 * w + 1) * 32 + kt, lane), a1);
      }
#pragma unroll
      for (int r = 0; r < 8; ++r) {
        const int m = r + (hi16 << 3);
        x_lds[m * LAT_ + w * 32 + nloc]      = f2bf(a0[r]);
        x_lds[m * LAT_ + w * 32 + 16 + nloc] = f2bf(a1[r]);
      }
    }

    // emit out[:, t, 0] and reset accumulator
    if (w == 0 && lane < 16) {
      out[(size_t)(b0 + lane) * T + t] = out_acc[lane] + bfc1;
      out_acc[lane] = 0.0f;
    }

    __syncthreads();
    buf ^= 1;
  }
}

extern "C" void kernel_launch(void* const* d_in, const int* in_sizes, int n_in,
                              void* d_out, int out_size, void* d_ws, size_t ws_size,
                              hipStream_t stream) {
  const float* x0    = (const float*)d_in[0];
  // d_in[1] = lengths: T is static in the reference -> recover as out_size / BS
  const float* Wih_f = (const float*)d_in[2];
  const float* Whh_f = (const float*)d_in[3];
  const float* b_f   = (const float*)d_in[4];
  const float* Wih_b = (const float*)d_in[5];
  const float* Whh_b = (const float*)d_in[6];
  const float* b_b   = (const float*)d_in[7];
  const float* W_out = (const float*)d_in[8];
  const float* b_out = (const float*)d_in[9];
  const float* W_fc1 = (const float*)d_in[10];
  const float* b_fc1 = (const float*)d_in[11];

  unsigned short* ws = (unsigned short*)d_ws;   // bf16 packed weights (25 MB)
  unsigned short* pk_ih_f = ws;                 // 4096x512
  unsigned short* pk_hh_f = ws + 2097152;       // 4096x1024
  unsigned short* pk_ih_b = ws + 6291456;       // 4096x512
  unsigned short* pk_hh_b = ws + 8388608;       // 4096x1024
  unsigned short* pk_out  = ws + 12582912;      // 512x1024

  const int T = out_size / BS_;

  dim3 blk(256);
  pack_w<<<(4096 * 512  / 16 + 255) / 256, blk, 0, stream>>>(Wih_f, pk_ih_f, 4096, 512);
  pack_w<<<(4096 * 1024 / 16 + 255) / 256, blk, 0, stream>>>(Whh_f, pk_hh_f, 4096, 1024);
  pack_w<<<(4096 * 512  / 16 + 255) / 256, blk, 0, stream>>>(Wih_b, pk_ih_b, 4096, 512);
  pack_w<<<(4096 * 1024 / 16 + 255) / 256, blk, 0, stream>>>(Whh_b, pk_hh_b, 4096, 1024);
  pack_w<<<(512  * 1024 / 16 + 255) / 256, blk, 0, stream>>>(W_out, pk_out, 512, 1024);

  if (T > 0)
    lstm_dec<<<8, 512, 0, stream>>>(x0, b_f, b_b, b_out, W_fc1, b_fc1,
                                    pk_ih_f, pk_hh_f, pk_ih_b, pk_hh_b, pk_out,
                                    (float*)d_out, T);
}
